// ErnieRnaSelfAttention_17325898072609
// MI455X (gfx1250) — matrix operand order repaired and checked by reading, not compiled, not run on hardware
//
#include <hip/hip_runtime.h>
#include <hip/hip_bf16.h>

typedef __attribute__((ext_vector_type(16))) _Float16 v16h;
typedef __attribute__((ext_vector_type(8)))  float    v8f;

#define B_   4
#define L_   1024
#define HID_ 768
#define H_   12
#define D_   64
#define SCALING_ 0.125f
// c = (l0+i) - (r0+j) + 1025 = (l0 - r0 + 962) + (i - j + 63);  t = i-j+63 in [0,126]
#define OFFC 962

__device__ inline v8f wmma16(v16h a, v16h b, v8f c) {
  return __builtin_amdgcn_wmma_f32_16x16x32_f16(false, a, false, b, (short)0, c,
                                                false, false);
}

// ---- CDNA5 async global->LDS copy (ASYNCcnt path, ISA 08 §4) ----------------
__device__ inline void async_copy_b128(void* lds_ptr, const void* gptr) {
  // flat->LDS aperture: low 32 bits of a generic LDS pointer are the LDS offset
  const unsigned lds = (unsigned)(unsigned long long)lds_ptr;
  asm volatile("global_load_async_to_lds_b128 %0, %1, off"
               :
               : "v"(lds), "v"((unsigned long long)gptr)
               : "memory");
}
__device__ inline void wait_async() {
  asm volatile("s_wait_asynccnt 0x0" ::: "memory");
}

// pack 8 f32 -> 8 f16 (16B) for vectorized LDS staging
union pk8 { _Float16 h[8]; uint4 u; };
union pk4 { _Float16 h[4]; uint2 u; };

// A fragment: logical 16x32 (MxK) f16, source stored row-major [m][k] in LDS.
__device__ inline v16h load_A_frag(const _Float16* p, int ld, int m0, int k0) {
  const int lane = threadIdx.x & 31;
  const int m = m0 + (lane & 15);
  const int g = lane >> 4;
  v16h a;
#pragma unroll
  for (int e = 0; e < 16; ++e) {
    const int k = g * 16 + e;
    a[e] = p[m * ld + k0 + k];
  }
  return a;
}

// B fragment: logical 32x16 (KxN) f16, source stored as [n][k].
__device__ inline v16h load_B_frag_nk(const _Float16* p, int ld, int k0, int n0) {
  const int lane = threadIdx.x & 31;
  const int n = n0 + (lane & 15);
  const int g = lane >> 4;
  v16h b;
#pragma unroll
  for (int e = 0; e < 16; ++e) b[e] = p[n * ld + k0 + g * 16 + e];
  return b;
}

// B fragment when source is stored as [k][n] (e.g. V tile [r][d]).
__device__ inline v16h load_B_frag_kn(const _Float16* p, int ld, int k0, int n0) {
  const int lane = threadIdx.x & 31;
  const int n = n0 + (lane & 15);
  const int g = lane >> 4;
  v16h b;
#pragma unroll
  for (int e = 0; e < 16; ++e) b[e] = p[(k0 + g * 16 + e) * ld + n];
  return b;
}

// C/D accumulator layout: N = lane%16, M = vgpr + 8*(lane/16).
__device__ inline void store_acc(float* p, int ld, int m0, int n0, v8f acc) {
  const int lane = threadIdx.x & 31;
  const int n = n0 + (lane & 15);
  const int g = lane >> 4;
#pragma unroll
  for (int v = 0; v < 8; ++v) p[(m0 + v + 8 * g) * ld + n] = acc[v];
}

// ---------------------------------------------------------------------------
// Kernel A: q/k/v projection.  out[m,n] = sum_k hs[m,k]*W[n,k] + bias[n]
// ---------------------------------------------------------------------------
__global__ void __launch_bounds__(256)
qkv_proj_kernel(const float* __restrict__ hs, const float* __restrict__ W,
                const float* __restrict__ bias, _Float16* __restrict__ out) {
  __shared__ _Float16 As[64 * 72];
  __shared__ _Float16 Bs[64 * 72];
  const int tid = threadIdx.x;
  const int wave = tid >> 5;
  const int m0 = blockIdx.y * 64;
  const int n0 = blockIdx.x * 64;
  const int msub = (wave >> 1) * 16;
  const int nsub = (wave & 1) * 32;
  v8f acc0 = {};
  v8f acc1 = {};
  for (int k0 = 0; k0 < HID_; k0 += 64) {
#pragma unroll
    for (int e = 0; e < 2; ++e) {            // 512 chunks of 8 elems
      const int cid = e * 256 + tid;
      const int r = cid >> 3, c8 = (cid & 7) * 8;
      const float4 a0 = *(const float4*)&hs[(size_t)(m0 + r) * HID_ + k0 + c8];
      const float4 a1 = *(const float4*)&hs[(size_t)(m0 + r) * HID_ + k0 + c8 + 4];
      pk8 pa;
      pa.h[0] = (_Float16)a0.x; pa.h[1] = (_Float16)a0.y;
      pa.h[2] = (_Float16)a0.z; pa.h[3] = (_Float16)a0.w;
      pa.h[4] = (_Float16)a1.x; pa.h[5] = (_Float16)a1.y;
      pa.h[6] = (_Float16)a1.z; pa.h[7] = (_Float16)a1.w;
      *(uint4*)&As[r * 72 + c8] = pa.u;
      const float4 b0 = *(const float4*)&W[(size_t)(n0 + r) * HID_ + k0 + c8];
      const float4 b1 = *(const float4*)&W[(size_t)(n0 + r) * HID_ + k0 + c8 + 4];
      pk8 pb;
      pb.h[0] = (_Float16)b0.x; pb.h[1] = (_Float16)b0.y;
      pb.h[2] = (_Float16)b0.z; pb.h[3] = (_Float16)b0.w;
      pb.h[4] = (_Float16)b1.x; pb.h[5] = (_Float16)b1.y;
      pb.h[6] = (_Float16)b1.z; pb.h[7] = (_Float16)b1.w;
      *(uint4*)&Bs[r * 72 + c8] = pb.u;
    }
    __syncthreads();
#pragma unroll
    for (int ks = 0; ks < 64; ks += 32) {
      v16h a  = load_A_frag(As, 72, msub, ks);
      v16h b0 = load_B_frag_nk(Bs, 72, ks, nsub);
      v16h b1 = load_B_frag_nk(Bs, 72, ks, nsub + 16);
      acc0 = wmma16(a, b0, acc0);
      acc1 = wmma16(a, b1, acc1);
    }
    __syncthreads();
  }
  const int lane = tid & 31;
  const int g = lane >> 4;
  const int ncol = lane & 15;
#pragma unroll
  for (int v = 0; v < 8; ++v) {
    const int m = m0 + msub + v + 8 * g;
    const int bb = m >> 10, l = m & 1023;
    {
      const int n = n0 + nsub + ncol;
      const int h = n >> 6, d = n & 63;
      out[(((size_t)bb * H_ + h) * L_ + l) * D_ + d] = (_Float16)(acc0[v] + bias[n]);
    }
    {
      const int n = n0 + nsub + 16 + ncol;
      const int h = n >> 6, d = n & 63;
      out[(((size_t)bb * H_ + h) * L_ + l) * D_ + d] = (_Float16)(acc1[v] + bias[n]);
    }
  }
}

// ---------------------------------------------------------------------------
// Kernel B: fused rel-pos scores + softmax + P@V.
// One WG = (b, h, 64 rows of l).  8 waves, 256 threads, ~129 KB dynamic LDS.
// ---------------------------------------------------------------------------
__global__ void __launch_bounds__(256)
attn_kernel(const _Float16* __restrict__ qws, const _Float16* __restrict__ kws,
            const _Float16* __restrict__ vws, const float* __restrict__ mask,
            const float* __restrict__ abias, const float* __restrict__ distemb,
            float* __restrict__ out_attn, float* __restrict__ out_scores,
            float* __restrict__ out_weights) {
  extern __shared__ char smem[];
  _Float16* Qb = (_Float16*)smem;       // 64 x 72
  _Float16* Kt = Qb + 64 * 72;          // 64 x 72 (reused as P tile in pass 2b)
  _Float16* Vt = Kt + 64 * 72;          // 64 x 72
  _Float16* Ew = Vt + 64 * 72;          // 128 x 72
  float* G1  = (float*)(Ew + 128 * 72); // 64 x 132
  float* G2  = G1 + 64 * 132;           // 64 x 132
  float* Sl  = G2 + 64 * 132;           // 64 x 68
  float* red = Sl + 64 * 68;            // 64 x 4

  const int tid  = threadIdx.x;
  const int wave = tid >> 5;
  const int l0 = blockIdx.x * 64;
  const int h  = blockIdx.y;
  const int b  = blockIdx.z;
  const size_t qkvbase = (((size_t)b * H_ + h) * L_) * D_;
  const size_t sbase   = (((size_t)b * H_ + h) * L_ + l0) * L_;

  // stage Q block via async DMA to LDS (512 x 16B chunks, 2 per thread)
#pragma unroll
  for (int e = 0; e < 2; ++e) {
    const int cid = e * 256 + tid;
    const int r = cid >> 3, c8 = (cid & 7) * 8;
    async_copy_b128(&Qb[r * 72 + c8], &qws[qkvbase + (size_t)(l0 + r) * D_ + c8]);
  }

  const int msub = (wave >> 1) * 16;
  const int nsub = (wave & 1) * 32;
  const int irow = tid >> 2;  // element-pass row within the 64-row block
  const int cgrp = tid & 3;   // 16-wide column chunk

  float mloc = -3.0e38f;

  // -------------------- pass 1: scores + row max --------------------
  for (int rt = 0; rt < 16; ++rt) {
    const int r0 = rt * 64;
    __syncthreads();  // previous tile's LDS consumers done
#pragma unroll
    for (int e = 0; e < 2; ++e) {
      const int cid = e * 256 + tid;
      const int r = cid >> 3, c8 = (cid & 7) * 8;
      async_copy_b128(&Kt[r * 72 + c8], &kws[qkvbase + (size_t)(r0 + r) * D_ + c8]);
    }
    const int cbase = l0 - r0 + OFFC;  // always in-bounds: c in [2, 2049]
#pragma unroll
    for (int e = 0; e < 4; ++e) {      // 1024 chunks of 8 elems
      const int cid = e * 256 + tid;
      const int t = cid >> 3, c8 = (cid & 7) * 8;
      const float* src = &distemb[(size_t)(cbase + t) * D_ + c8];
      const float4 f0 = *(const float4*)src;
      const float4 f1 = *(const float4*)(src + 4);
      pk8 pe;
      pe.h[0] = (_Float16)f0.x; pe.h[1] = (_Float16)f0.y;
      pe.h[2] = (_Float16)f0.z; pe.h[3] = (_Float16)f0.w;
      pe.h[4] = (_Float16)f1.x; pe.h[5] = (_Float16)f1.y;
      pe.h[6] = (_Float16)f1.z; pe.h[7] = (_Float16)f1.w;
      *(uint4*)&Ew[t * 72 + c8] = pe.u;
    }
    wait_async();
    __syncthreads();

    // S = Q @ K^T  (this wave's 16x32 area)
    {
      v8f s0 = {};
      v8f s1 = {};
#pragma unroll
      for (int ks = 0; ks < 64; ks += 32) {
        v16h a  = load_A_frag(Qb, 72, msub, ks);
        v16h b0 = load_B_frag_nk(Kt, 72, ks, nsub);
        v16h b1 = load_B_frag_nk(Kt, 72, ks, nsub + 16);
        s0 = wmma16(a, b0, s0);
        s1 = wmma16(a, b1, s1);
      }
      store_acc(Sl, 68, msub, nsub, s0);
      store_acc(Sl, 68, msub, nsub + 16, s1);
    }
    // G1 = Q @ Ewin^T, G2 = K @ Ewin^T  (64 x 128 each, Toeplitz window)
    {
      const int trow = msub;
      const int tc0 = (wave & 1) * 64;
#pragma unroll
      for (int s = 0; s < 4; ++s) {
        const int tc = tc0 + s * 16;
        v8f g1 = {};
        v8f g2 = {};
#pragma unroll
        for (int ks = 0; ks < 64; ks += 32) {
          v16h be = load_B_frag_nk(Ew, 72, ks, tc);
          v16h aq = load_A_frag(Qb, 72, trow, ks);
          v16h ak = load_A_frag(Kt, 72, trow, ks);
          g1 = wmma16(aq, be, g1);
          g2 = wmma16(ak, be, g2);
        }
        store_acc(G1, 132, trow, tc, g1);
        store_acc(G2, 132, trow, tc, g2);
      }
    }
    __syncthreads();

    // element pass: scores = SCALING*(qk + rel1 + rel2) + mask + bias
#pragma unroll
    for (int q4 = 0; q4 < 4; ++q4) {
      const int j0 = cgrp * 16 + q4 * 4;
      const int r = r0 + j0;
      const float4 bi = *(const float4*)&abias[sbase + (size_t)irow * L_ + r];
      const float4 mk = *(const float4*)&mask[b * L_ + r];
      float4 sv;
      {
        const int t = irow - j0 + 63;
        sv.x = SCALING_ * (Sl[irow*68 + j0    ] + G1[irow*132 + t    ] + G2[(j0    )*132 + t    ]) + mk.x + bi.x;
        sv.y = SCALING_ * (Sl[irow*68 + j0 + 1] + G1[irow*132 + t - 1] + G2[(j0 + 1)*132 + t - 1]) + mk.y + bi.y;
        sv.z = SCALING_ * (Sl[irow*68 + j0 + 2] + G1[irow*132 + t - 2] + G2[(j0 + 2)*132 + t - 2]) + mk.z + bi.z;
        sv.w = SCALING_ * (Sl[irow*68 + j0 + 3] + G1[irow*132 + t - 3] + G2[(j0 + 3)*132 + t - 3]) + mk.w + bi.w;
      }
      *(float4*)&out_scores[sbase + (size_t)irow * L_ + r] = sv;
      mloc = fmaxf(mloc, fmaxf(fmaxf(sv.x, sv.y), fmaxf(sv.z, sv.w)));
    }
  }
  __syncthreads();
  red[irow * 4 + cgrp] = mloc;
  __syncthreads();
  const float mrow = fmaxf(fmaxf(red[irow * 4 + 0], red[irow * 4 + 1]),
                           fmaxf(red[irow * 4 + 2], red[irow * 4 + 3]));

  // -------------------- pass 2a: row exp-sums (L2-hot re-read) --------------
  float lsum = 0.0f;
  for (int rt = 0; rt < 16; ++rt) {
    const int r0 = rt * 64;
#pragma unroll
    for (int q4 = 0; q4 < 4; ++q4) {
      const int r = r0 + cgrp * 16 + q4 * 4;
      const float4 sv = *(const float4*)&out_scores[sbase + (size_t)irow * L_ + r];
      lsum += __expf(sv.x - mrow) + __expf(sv.y - mrow) +
              __expf(sv.z - mrow) + __expf(sv.w - mrow);
    }
  }
  __syncthreads();
  red[irow * 4 + cgrp] = lsum;
  __syncthreads();
  const float linv = 1.0f / (red[irow * 4 + 0] + red[irow * 4 + 1] +
                             red[irow * 4 + 2] + red[irow * 4 + 3]);

  // -------------------- pass 2b: weights + O = P @ V --------------------
  v8f o0 = {};
  v8f o1 = {};
  _Float16* Pt = Kt;  // reuse K tile space for P (f16)
  for (int rt = 0; rt < 16; ++rt) {
    const int r0 = rt * 64;
    __syncthreads();
#pragma unroll
    for (int e = 0; e < 2; ++e) {
      const int cid = e * 256 + tid;
      const int r = cid >> 3, c8 = (cid & 7) * 8;
      async_copy_b128(&Vt[r * 72 + c8], &vws[qkvbase + (size_t)(r0 + r) * D_ + c8]);
    }
#pragma unroll
    for (int q4 = 0; q4 < 4; ++q4) {
      const int j0 = cgrp * 16 + q4 * 4;
      const int r = r0 + j0;
      const float4 sv = *(const float4*)&out_scores[sbase + (size_t)irow * L_ + r];
      float4 pv;
      pv.x = __expf(sv.x - mrow) * linv;
      pv.y = __expf(sv.y - mrow) * linv;
      pv.z = __expf(sv.z - mrow) * linv;
      pv.w = __expf(sv.w - mrow) * linv;
      *(float4*)&out_weights[sbase + (size_t)irow * L_ + r] = pv;
      pk4 pp;
      pp.h[0] = (_Float16)pv.x; pp.h[1] = (_Float16)pv.y;
      pp.h[2] = (_Float16)pv.z; pp.h[3] = (_Float16)pv.w;
      *(uint2*)&Pt[irow * 72 + j0] = pp.u;
    }
    wait_async();
    __syncthreads();
#pragma unroll
    for (int ks = 0; ks < 64; ks += 32) {
      v16h a  = load_A_frag(Pt, 72, msub, ks);
      v16h b0 = load_B_frag_kn(Vt, 72, ks, nsub);
      v16h b1 = load_B_frag_kn(Vt, 72, ks, nsub + 16);
      o0 = wmma16(a, b0, o0);
      o1 = wmma16(a, b1, o1);
    }
  }
  // write O: attn_output[b][l][h*64+d]
  {
    const int lane = tid & 31;
    const int g = lane >> 4;
    const int ncol = lane & 15;
#pragma unroll
    for (int v = 0; v < 8; ++v) {
      const int l = l0 + msub + v + 8 * g;
      out_attn[((size_t)b * L_ + l) * HID_ + h * D_ + nsub + ncol] = o0[v];
      out_attn[((size_t)b * L_ + l) * HID_ + h * D_ + nsub + 16 + ncol] = o1[v];
    }
  }
}

extern "C" void kernel_launch(void* const* d_in, const int* in_sizes, int n_in,
                              void* d_out, int out_size, void* d_ws, size_t ws_size,
                              hipStream_t stream) {
  (void)in_sizes; (void)n_in; (void)out_size; (void)ws_size;
  const float* hs    = (const float*)d_in[0];
  const float* mask  = (const float*)d_in[1];
  const float* abias = (const float*)d_in[2];
  const float* Wq = (const float*)d_in[3];
  const float* bq = (const float*)d_in[4];
  const float* Wk = (const float*)d_in[5];
  const float* bk = (const float*)d_in[6];
  const float* Wv = (const float*)d_in[7];
  const float* bv = (const float*)d_in[8];
  const float* de = (const float*)d_in[9];

  float* out_attn    = (float*)d_out;
  float* out_scores  = out_attn + (size_t)B_ * L_ * HID_;
  float* out_weights = out_scores + (size_t)B_ * H_ * L_ * L_;

  const size_t qkvelems = (size_t)B_ * H_ * L_ * D_;
  _Float16* qws = (_Float16*)d_ws;
  _Float16* kws = qws + qkvelems;
  _Float16* vws = kws + qkvelems;

  dim3 gA(HID_ / 64, (B_ * L_) / 64);
  qkv_proj_kernel<<<gA, 256, 0, stream>>>(hs, Wq, bq, qws);
  qkv_proj_kernel<<<gA, 256, 0, stream>>>(hs, Wk, bk, kws);
  qkv_proj_kernel<<<gA, 256, 0, stream>>>(hs, Wv, bv, vws);

  dim3 gB(L_ / 64, H_, B_);
  const size_t smem =
      (size_t)(64 * 72 * 3 + 128 * 72) * sizeof(_Float16) +
      (size_t)(64 * 132 * 2 + 64 * 68 + 64 * 4) * sizeof(float);
  attn_kernel<<<gB, 256, smem, stream>>>(qws, kws, vws, mask, abias, de,
                                         out_attn, out_scores, out_weights);
}
